// StepModelTransformer_12506944766386
// MI455X (gfx1250) — matrix operand order repaired and checked by reading, not compiled, not run on hardware
//
#include <hip/hip_runtime.h>
#include <hip/hip_bf16.h>

// ===========================================================================
// Snowflake point-completion forward for MI455X (gfx1250, wave32).
// All 1x1-conv / MLP math runs through one f16 WMMA GEMM kernel
// (v_wmma_f32_16x16x32_f16, f32 accumulate, fused bias/residual/act):
//   - 64x128 block tile, 8 waves, each wave 32x32 -> 4 WMMAs/K-step with
//     chained matrix_a/b reuse hints
//   - full-tile staging is unmasked batched loads; only the K tail is masked
//     (by multiply, so loads can't be predicated into serial exec regions)
//   - register double-buffering of K tiles; packed v_cvt_pk_rtz_f16_f32 LDS
//     commits; aligned ds_load_b128 fragment reads
// FPS keeps its distance array in LDS; KNN keeps a register top-16.
// ===========================================================================

typedef __attribute__((ext_vector_type(16))) _Float16 v16h;
typedef __attribute__((ext_vector_type(8)))  _Float16 v8h;
typedef __attribute__((ext_vector_type(2)))  __fp16   v2fp;   // cvt_pkrtz result type
typedef __attribute__((ext_vector_type(8)))  float    v8f;

static inline int cdiv(int a, int b) { return (a + b - 1) / b; }

// Stage one 32-deep K slice into registers. TAIL=false: unmasked batched
// loads (steady state). TAIL=true: clamped addresses + multiplicative mask.
template <bool TAIL>
__device__ __forceinline__
void stage_frag(const float* __restrict__ aRow, const float* __restrict__ bCol,
                int K, int L, int kbase, int ka, int kb, float mA, float mB,
                float (&ra)[8], float (&rb)[16]) {
#pragma unroll
  for (int j = 0; j < 8; ++j) {
    int gk = kbase + ka + j;
    int gks = TAIL ? (gk < K ? gk : K - 1) : gk;
    float m = TAIL ? ((gk < K) ? mA : 0.f) : mA;
    ra[j] = aRow[gks] * m;
  }
#pragma unroll
  for (int j = 0; j < 16; ++j) {
    int gk = kbase + kb + j;
    int gks = TAIL ? (gk < K ? gk : K - 1) : gk;
    float m = TAIL ? ((gk < K) ? mB : 0.f) : mB;
    rb[j] = bCol[(size_t)gks * L] * m;
  }
}

// ---------------------------------------------------------------------------
// WMMA GEMM:  Y[b](M x L) = act( W(M x K) * X[b](K x L) + bias + res[b] )
// Block tile 64(M) x 128(L), 8 waves (2x4), each wave 32x32 (4 wmma/step).
// act: 0 = none, 1 = relu, 2 = tanh
// ---------------------------------------------------------------------------
__global__ __launch_bounds__(256)
void k_gemm_wmma(const float* __restrict__ W, const float* __restrict__ X,
                 const float* __restrict__ bias, const float* res,
                 float* Y, int M, int K, int L, int act)
{
  __shared__ _Float16 sA[64][32];    // [m][k]
  __shared__ _Float16 sB[128][32];   // [n][k]
  const int b  = blockIdx.z;
  const float* Xb = X + (size_t)b * K * L;
  float*       Yb = Y + (size_t)b * M * L;
  const float* Rb = res ? res + (size_t)b * M * L : nullptr;
  const int m0 = blockIdx.y * 64, n0 = blockIdx.x * 128;
  const int tid  = threadIdx.x;
  const int wave = tid >> 5, lane = tid & 31;
  const int wm = wave >> 2, wn = wave & 3;   // 2 x 4 wave grid, 32x32 each
  const int lm = lane & 15, hi = lane >> 4;

  // staging coordinates
  const int ma = tid >> 2;                 // 0..63  : A row within tile
  const int ka = (tid & 3) * 8;            // 0..24  : 8 consecutive k
  const int gmA = m0 + ma;
  const float mA = (gmA < M) ? 1.f : 0.f;
  const float* aRow = W + (size_t)(gmA < M ? gmA : M - 1) * K;

  const int nb = tid & 127;                // 0..127 : B col within tile
  const int kb = (tid >> 7) * 16;          // 0 or 16: 16 consecutive k
  const int gnB = n0 + nb;
  const float mB = (gnB < L) ? 1.f : 0.f;
  const float* bCol = Xb + (gnB < L ? gnB : L - 1);

  float ra[8], rb[16];
  auto stage = [&](int kbase) {
    if (kbase + 32 <= K)
      stage_frag<false>(aRow, bCol, K, L, kbase, ka, kb, mA, mB, ra, rb);
    else
      stage_frag<true>(aRow, bCol, K, L, kbase, ka, kb, mA, mB, ra, rb);
  };
  stage(0);

  v8f acc[2][2] = {};
  for (int k0 = 0; k0 < K; k0 += 32) {
    // commit staged registers to LDS as packed f16 pairs
#pragma unroll
    for (int j = 0; j < 8; j += 2)
      *(v2fp*)&sA[ma][ka + j] = __builtin_amdgcn_cvt_pkrtz(ra[j], ra[j + 1]);
#pragma unroll
    for (int j = 0; j < 16; j += 2)
      *(v2fp*)&sB[nb][kb + j] = __builtin_amdgcn_cvt_pkrtz(rb[j], rb[j + 1]);
    __syncthreads();
    // prefetch next K tile into registers while the matrix pipe works
    if (k0 + 32 < K) stage(k0 + 32);
    // fragments: A row = lm; lanes 0-15 hold K {0..7,16..23}, 16-31 the rest
    v16h af[2], bf[2];
#pragma unroll
    for (int t1 = 0; t1 < 2; ++t1) {
      v8h a0 = *(const v8h*)&sA[wm * 32 + t1 * 16 + lm][hi * 8];
      v8h a1 = *(const v8h*)&sA[wm * 32 + t1 * 16 + lm][16 + hi * 8];
      af[t1] = __builtin_shufflevector(a0, a1, 0, 1, 2, 3, 4, 5, 6, 7,
                                       8, 9, 10, 11, 12, 13, 14, 15);
      v8h b0 = *(const v8h*)&sB[wn * 32 + t1 * 16 + lm][hi * 8];
      v8h b1 = *(const v8h*)&sB[wn * 32 + t1 * 16 + lm][16 + hi * 8];
      bf[t1] = __builtin_shufflevector(b0, b1, 0, 1, 2, 3, 4, 5, 6, 7,
                                       8, 9, 10, 11, 12, 13, 14, 15);
    }
    acc[0][0] = __builtin_amdgcn_wmma_f32_16x16x32_f16(false, af[0], false, bf[0],
                                                       (short)0, acc[0][0], false, false);
    acc[0][1] = __builtin_amdgcn_wmma_f32_16x16x32_f16(false, af[0], false, bf[1],
                                                       (short)0, acc[0][1], true, false);
    acc[1][1] = __builtin_amdgcn_wmma_f32_16x16x32_f16(false, af[1], false, bf[1],
                                                       (short)0, acc[1][1], false, true);
    acc[1][0] = __builtin_amdgcn_wmma_f32_16x16x32_f16(false, af[1], false, bf[0],
                                                       (short)0, acc[1][0], true, false);
    __syncthreads();
  }
  // epilogue: C/D layout vgpr r -> M = r + 8*hi, N = lm
#pragma unroll
  for (int t1 = 0; t1 < 2; ++t1) {
    const int rowB = m0 + wm * 32 + t1 * 16;
#pragma unroll
    for (int t2 = 0; t2 < 2; ++t2) {
      int col = n0 + wn * 32 + t2 * 16 + lm;
      if (col < L) {
#pragma unroll
        for (int r = 0; r < 8; ++r) {
          int row = rowB + r + 8 * hi;
          if (row < M) {
            float v = acc[t1][t2][r];
            if (bias) v += bias[row];
            if (Rb)   v += Rb[(size_t)row * L + col];
            if (act == 1)      v = v > 0.f ? v : 0.f;
            else if (act == 2) v = tanhf(v);
            Yb[(size_t)row * L + col] = v;
          }
        }
      }
    }
  }
}

// ---------------------------------------------------------------------------
// Farthest point sampling: one workgroup per batch, distance array in LDS.
// xyz (B,N,3) row-major, out (B,S) int32. Matches reference (start idx 0).
// ---------------------------------------------------------------------------
__global__ __launch_bounds__(256)
void k_fps(const float* __restrict__ xyz, int* __restrict__ out, int N, int S)
{
  __shared__ float dist[2304];
  __shared__ float rv[256];
  __shared__ int   ri[256];
  __shared__ float cx, cy, cz;
  __shared__ int   cur;
  const int b = blockIdx.x, tid = threadIdx.x;
  const float* P = xyz + (size_t)b * N * 3;
  for (int i = tid; i < N; i += 256) dist[i] = 1e10f;
  if (tid == 0) cur = 0;
  __syncthreads();
  for (int s = 0; s < S; ++s) {
    if (tid == 0) {
      out[b * S + s] = cur;
      cx = P[cur * 3]; cy = P[cur * 3 + 1]; cz = P[cur * 3 + 2];
    }
    __syncthreads();
    float bm = -1.f; int bi = 0;
    for (int i = tid; i < N; i += 256) {
      float dx = P[i * 3] - cx, dy = P[i * 3 + 1] - cy, dz = P[i * 3 + 2] - cz;
      float nd = fminf(dist[i], dx * dx + dy * dy + dz * dz);
      dist[i] = nd;
      if (nd > bm) { bm = nd; bi = i; }
    }
    rv[tid] = bm; ri[tid] = bi;
    __syncthreads();
    for (int o = 128; o > 0; o >>= 1) {
      if (tid < o && rv[tid + o] > rv[tid]) { rv[tid] = rv[tid + o]; ri[tid] = ri[tid + o]; }
      __syncthreads();
    }
    if (tid == 0) cur = ri[0];
    __syncthreads();
  }
}

// ---------------------------------------------------------------------------
// Brute-force 16-NN, register-resident sorted list. src/qry (B,*,3) rows.
// ---------------------------------------------------------------------------
__global__ void k_knn(const float* __restrict__ src, const float* __restrict__ qry,
                      int* __restrict__ idx, int N, int M)
{
  const int b = blockIdx.y;
  const int m = blockIdx.x * blockDim.x + threadIdx.x;
  if (m >= M) return;
  const float* S = src + (size_t)b * N * 3;
  const float* Q = qry + (size_t)b * M * 3;
  const float qx = Q[m * 3], qy = Q[m * 3 + 1], qz = Q[m * 3 + 2];
  float bd[16]; int bi[16];
#pragma unroll
  for (int j = 0; j < 16; ++j) { bd[j] = 3.0e38f; bi[j] = 0; }
  for (int n = 0; n < N; ++n) {
    __builtin_prefetch(S + (n + 128) * 3, 0, 1);   // global_prefetch_b8
    float dx = S[n * 3] - qx, dy = S[n * 3 + 1] - qy, dz = S[n * 3 + 2] - qz;
    float d = dx * dx + dy * dy + dz * dz;
    if (d < bd[15]) {
      bd[15] = d; bi[15] = n;
#pragma unroll
      for (int j = 15; j > 0; --j) {
        if (bd[j] < bd[j - 1]) {
          float td = bd[j]; bd[j] = bd[j - 1]; bd[j - 1] = td;
          int   ti = bi[j]; bi[j] = bi[j - 1]; bi[j - 1] = ti;
        }
      }
    }
  }
  int* O = idx + ((size_t)b * M + m) * 16;
#pragma unroll
  for (int j = 0; j < 16; ++j) O[j] = bi[j];
}

// ---------------- light service kernels (tot is always the last arg) -------
__global__ void k_perm021(const float* __restrict__ in, float* __restrict__ out,
                          int D1, int D2, size_t tot) {
  size_t i = (size_t)blockIdx.x * blockDim.x + threadIdx.x;
  if (i >= tot) return;
  int d2 = (int)(i % D2); size_t r = i / D2;
  int d1 = (int)(r % D1); int b = (int)(r / D1);
  out[((size_t)b * D2 + d2) * D1 + d1] = in[i];
}

__global__ void k_gather_ch(const float* __restrict__ p, const int* __restrict__ idx,
                            float* __restrict__ out, int C, int N, int M, size_t tot) {
  size_t i = (size_t)blockIdx.x * blockDim.x + threadIdx.x;
  if (i >= tot) return;
  int m = (int)(i % M); size_t r = i / M;
  int c = (int)(r % C); int b = (int)(r / C);
  out[i] = p[((size_t)b * C + c) * N + idx[(size_t)b * M + m]];
}

__global__ void k_gather_rows3(const float* __restrict__ in, const int* __restrict__ idx,
                               float* __restrict__ out, int N, int S, size_t tot) {
  size_t i = (size_t)blockIdx.x * blockDim.x + threadIdx.x;
  if (i >= tot) return;
  int d = (int)(i % 3); size_t r = i / 3;
  int s = (int)(r % S); int b = (int)(r / S);
  out[i] = in[((size_t)b * N + idx[(size_t)b * S + s]) * 3 + d];
}

__global__ void k_copy_rows3(const float* __restrict__ in, float* __restrict__ out,
                             int R, int Rt, int roff, size_t tot) {
  size_t i = (size_t)blockIdx.x * blockDim.x + threadIdx.x;
  if (i >= tot) return;
  int d = (int)(i % 3); size_t r2 = i / 3;
  int r = (int)(r2 % R); int b = (int)(r2 / R);
  out[((size_t)b * Rt + roff + r) * 3 + d] = in[i];
}

// group (optionally minus per-center q) into a channel-offset slot of a
// larger (B,Cd,M,16) tensor.
__global__ void k_group16(const float* __restrict__ p, const int* __restrict__ idx,
                          const float* q, float* __restrict__ out,
                          int C, int N, int M, int Cd, int coff, size_t tot) {
  size_t i = (size_t)blockIdx.x * blockDim.x + threadIdx.x;
  if (i >= tot) return;
  int k = (int)(i & 15); size_t r = i >> 4;
  int m = (int)(r % M); r /= M;
  int c = (int)(r % C); int b = (int)(r / C);
  float v = p[((size_t)b * C + c) * N + idx[((size_t)b * M + m) * 16 + k]];
  if (q) v -= q[((size_t)b * C + c) * M + m];
  out[(((size_t)b * Cd + coff + c) * M + m) * 16 + k] = v;
}

// train-mode BN stats over (B, L) per channel; one block per channel.
__global__ __launch_bounds__(256)
void k_bn_stats(const float* __restrict__ x, float* __restrict__ mean,
                float* __restrict__ rstd, int B, int C, int L) {
  __shared__ float s1[256], s2[256];
  const int c = blockIdx.x, tid = threadIdx.x;
  float a = 0.f, q = 0.f;
  for (int b = 0; b < B; ++b) {
    const float* p = x + ((size_t)b * C + c) * L;
    for (int l = tid; l < L; l += 256) { float v = p[l]; a += v; q += v * v; }
  }
  s1[tid] = a; s2[tid] = q;
  __syncthreads();
  for (int o = 128; o > 0; o >>= 1) {
    if (tid < o) { s1[tid] += s1[tid + o]; s2[tid] += s2[tid + o]; }
    __syncthreads();
  }
  if (tid == 0) {
    float n = (float)B * (float)L;
    float m = s1[0] / n;
    float v = s2[0] / n - m * m;
    mean[c] = m; rstd[c] = rsqrtf(v + 1e-5f);
  }
}

__global__ void k_bn_relu(float* x, const float* __restrict__ mean,
                          const float* __restrict__ rstd, const float* __restrict__ g,
                          const float* __restrict__ bta, int C, int L, size_t tot) {
  size_t i = (size_t)blockIdx.x * blockDim.x + threadIdx.x;
  if (i >= tot) return;
  int c = (int)((i / L) % C);
  float v = g[c] * (x[i] - mean[c]) * rstd[c] + bta[c];
  x[i] = v > 0.f ? v : 0.f;
}

__global__ void k_bias_ch(float* x, const float* __restrict__ bb, int C, int L, size_t tot) {
  size_t i = (size_t)blockIdx.x * blockDim.x + threadIdx.x;
  if (i >= tot) return;
  x[i] += bb[(int)((i / L) % C)];
}

__global__ void k_qmk(const float* __restrict__ q, const float* __restrict__ kg,
                      const float* __restrict__ pe, float* __restrict__ h, size_t tot) {
  size_t i = (size_t)blockIdx.x * blockDim.x + threadIdx.x;
  if (i >= tot) return;
  h[i] = q[i >> 4] - kg[i] + pe[i];
}

__global__ void k_softmax16(float* x, size_t rows) {
  size_t i = (size_t)blockIdx.x * blockDim.x + threadIdx.x;
  if (i >= rows) return;
  float* p = x + i * 16;
  float mx = p[0];
#pragma unroll
  for (int j = 1; j < 16; ++j) mx = fmaxf(mx, p[j]);
  float s = 0.f;
#pragma unroll
  for (int j = 0; j < 16; ++j) { float e = expf(p[j] - mx); p[j] = e; s += e; }
  float inv = 1.f / s;
#pragma unroll
  for (int j = 0; j < 16; ++j) p[j] *= inv;
}

__global__ void k_attn_sum(const float* __restrict__ a, const float* __restrict__ v,
                           const float* __restrict__ pe, float* __restrict__ out, size_t rows) {
  size_t i = (size_t)blockIdx.x * blockDim.x + threadIdx.x;
  if (i >= rows) return;
  const float* A = a + i * 16; const float* V = v + i * 16; const float* P = pe + i * 16;
  float s = 0.f;
#pragma unroll
  for (int j = 0; j < 16; ++j) s += A[j] * (V[j] + P[j]);
  out[i] = s;
}

__global__ void k_maxlast(const float* __restrict__ x, float* __restrict__ out,
                          int Kk, size_t rows) {
  size_t i = (size_t)blockIdx.x * blockDim.x + threadIdx.x;
  if (i >= rows) return;
  const float* p = x + i * Kk;
  float m = p[0];
  for (int j = 1; j < Kk; ++j) m = fmaxf(m, p[j]);
  out[i] = m;
}

// copy / broadcast / repeat a (B,C,n) tensor into channel slot [coff,coff+C)
// of a (B,Cd,n*uf) tensor.  (uf=1: copy; n=1: broadcast; uf>1: repeat.)
__global__ void k_repeat_ch(const float* __restrict__ in, float* __restrict__ out,
                            int C, int n, int uf, int Cd, int coff, size_t tot) {
  size_t i = (size_t)blockIdx.x * blockDim.x + threadIdx.x;
  if (i >= tot) return;
  int Lo = n * uf;
  int t = (int)(i % Lo); size_t r = i / Lo;
  int c = (int)(r % C); int b = (int)(r / C);
  out[((size_t)b * Cd + coff + c) * Lo + t] = in[((size_t)b * C + c) * n + t / uf];
}

__global__ void k_repeat_add(const float* __restrict__ prev, const float* __restrict__ delta,
                             float* __restrict__ out, int n, int uf, size_t tot) {
  size_t i = (size_t)blockIdx.x * blockDim.x + threadIdx.x;
  if (i >= tot) return;
  int Lo = n * uf;
  int t = (int)(i % Lo); size_t r = i / Lo;
  int c = (int)(r % 3); int b = (int)(r / 3);
  out[i] = prev[((size_t)b * 3 + c) * n + t / uf] + delta[i];
}

__global__ void k_relu_copy(const float* __restrict__ x, float* __restrict__ y, size_t tot) {
  size_t i = (size_t)blockIdx.x * blockDim.x + threadIdx.x;
  if (i >= tot) return;
  float v = x[i];
  y[i] = v > 0.f ? v : 0.f;
}

// ConvTranspose1d(32,128,uf,stride=uf): out[b, coff+c, t*uf+j], K=32
__global__ void k_ps_deconv(const float* __restrict__ fc, const float* __restrict__ psw,
                            float* __restrict__ out, int n, int uf, int Cd, int coff, size_t tot) {
  size_t i = (size_t)blockIdx.x * blockDim.x + threadIdx.x;
  if (i >= tot) return;
  int Lo = n * uf;
  int t = (int)(i % Lo); size_t r = i / Lo;
  int c = (int)(r % 128); int b = (int)(r / 128);
  int tt = t / uf, j = t % uf;
  float s = 0.f;
#pragma unroll
  for (int k = 0; k < 32; ++k)
    s += fc[((size_t)b * 32 + k) * n + tt] * psw[((size_t)k * 128 + c) * uf + j];
  out[((size_t)b * Cd + coff + c) * Lo + t] = s;
}

// ===========================================================================
// Host-side orchestration
// ===========================================================================
#define L1D(kern, tot, ...) do { size_t _t = (size_t)(tot); \
  kern<<<dim3((unsigned)((_t + 255) / 256)), dim3(256), 0, st>>>(__VA_ARGS__, _t); } while (0)

namespace {

struct Arena {
  char* base; size_t off; size_t cap;
  float* f(size_t n) { size_t a = (off + 255) & ~(size_t)255; off = a + n * 4; return (float*)(base + a); }
  int*   i(size_t n) { size_t a = (off + 255) & ~(size_t)255; off = a + n * 4; return (int*)(base + a); }
};

struct MResW  { const float *b1, *b2, *bs, *w1, *w2, *ws; int ci, h, o; };
struct MConvW { const float *b0, *b1, *w0, *w1; int ci, h, o; };
struct AttnW  { const float *ab0, *ab1, *abt0, *ag0, *aw0, *aw1,
                            *pb0, *pb1, *pbt0, *pg0, *pw0, *pw1; };
struct SkipW  { AttnW a; const float *eb, *ew, *kb, *kw; MResW mlp_v;
                const float *qb, *qw, *vb, *vw; };
struct TransW { AttnW a; const float *kb, *kw, *leb, *lew, *lsb, *lsw,
                            *qb, *qw, *vb, *vw; int inC; };
struct SAW    { const float *b0, *b1, *w0, *w1; int ci, h, o; };
struct SpdW   { MConvW md, mlp1, mlp2, mlp_ps; MResW mdf; const float* psw;
                SkipW skip; int uf; };
struct SeedW  { const float *b40, *b41, *psb, *psw, *w40, *w41; MResW m1, m2, m3; };

void gemm(hipStream_t st, const float* W, const float* X, const float* bias,
          const float* res, float* Y, int B, int M, int K, int L, int act) {
  dim3 g((unsigned)cdiv(L, 128), (unsigned)cdiv(M, 64), (unsigned)B);
  k_gemm_wmma<<<g, 256, 0, st>>>(W, X, bias, res, Y, M, K, L, act);
}

void mlp_conv(hipStream_t st, Arena ar, const MConvW& w, const float* x, float* y,
              int B, int n, int act2) {
  float* t = ar.f((size_t)B * w.h * n);
  gemm(st, w.w0, x, w.b0, nullptr, t, B, w.h, w.ci, n, 1);
  gemm(st, w.w1, t, w.b1, nullptr, y, B, w.o, w.h, n, act2);
}

void mlp_res(hipStream_t st, Arena ar, const MResW& w, const float* x, float* y,
             int B, int n) {
  float* t = ar.f((size_t)B * w.h * n);
  float* s = ar.f((size_t)B * w.o * n);
  gemm(st, w.w1, x, w.b1, nullptr, t, B, w.h, w.ci, n, 1);
  gemm(st, w.ws, x, w.bs, nullptr, s, B, w.o, w.ci, n, 0);
  gemm(st, w.w2, t, w.b2, s, y, B, w.o, w.h, n, 0);
}

// pos/attn MLP core: dim=64, pos hidden 64, attn hidden 256, K=16
void attn_core(hipStream_t st, Arena ar, const AttnW& w, const float* q,
               const float* kg, const float* vg, const float* posrel,
               float* agg, int B, int M) {
  const int L = M * 16;
  const size_t e64 = (size_t)B * 64 * L, e256 = (size_t)B * 256 * L;
  float* t0 = ar.f(e64);
  gemm(st, w.pw0, posrel, w.pb0, nullptr, t0, B, 64, 3, L, 0);
  float* mu = ar.f(64); float* rs = ar.f(64);
  k_bn_stats<<<64, 256, 0, st>>>(t0, mu, rs, B, 64, L);
  L1D(k_bn_relu, e64, t0, mu, rs, w.pg0, w.pbt0, 64, L);
  float* pe = ar.f(e64);
  gemm(st, w.pw1, t0, w.pb1, nullptr, pe, B, 64, 64, L, 0);
  float* h = t0;                                   // reuse
  L1D(k_qmk, e64, q, kg, pe, h);
  float* ah = ar.f(e256);
  gemm(st, w.aw0, h, w.ab0, nullptr, ah, B, 256, 64, L, 0);
  float* mu2 = ar.f(256); float* rs2 = ar.f(256);
  k_bn_stats<<<256, 256, 0, st>>>(ah, mu2, rs2, B, 256, L);
  L1D(k_bn_relu, e256, ah, mu2, rs2, w.ag0, w.abt0, 256, L);
  float* a = h;                                    // reuse
  gemm(st, w.aw1, ah, w.ab1, nullptr, a, B, 64, 256, L, 0);
  const size_t rows = (size_t)B * 64 * M;
  L1D(k_softmax16, rows, a);
  L1D(k_attn_sum, rows, a, vg, pe, agg);
}

void transformer(hipStream_t st, Arena ar, const TransW& w, float* x,
                 const float* pos, int B, int M) {
  const int inC = w.inC;
  float* xs = ar.f((size_t)B * 64 * M);
  gemm(st, w.lsw, x, w.lsb, nullptr, xs, B, 64, inC, M, 0);
  float* posT = ar.f((size_t)B * M * 3);
  L1D(k_perm021, (size_t)B * 3 * M, pos, posT, 3, M);
  int* idx = ar.i((size_t)B * M * 16);
  k_knn<<<dim3((unsigned)cdiv(M, 128), (unsigned)B), 128, 0, st>>>(posT, posT, idx, M, M);
  float* kf = ar.f((size_t)B * 64 * M);
  float* kg = ar.f((size_t)B * 64 * M * 16);
  float* vg = ar.f((size_t)B * 64 * M * 16);
  gemm(st, w.kw, xs, w.kb, nullptr, kf, B, 64, 64, M, 0);
  L1D(k_group16, (size_t)B * 64 * M * 16, kf, idx, nullptr, kg, 64, M, M, 64, 0);
  gemm(st, w.vw, xs, w.vb, nullptr, kf, B, 64, 64, M, 0);
  L1D(k_group16, (size_t)B * 64 * M * 16, kf, idx, nullptr, vg, 64, M, M, 64, 0);
  float* q = kf;                                   // reuse
  gemm(st, w.qw, xs, w.qb, nullptr, q, B, 64, 64, M, 0);
  float* posrel = ar.f((size_t)B * 3 * M * 16);
  L1D(k_group16, (size_t)B * 3 * M * 16, pos, idx, pos, posrel, 3, M, M, 3, 0);
  float* agg = ar.f((size_t)B * 64 * M);
  attn_core(st, ar, w.a, q, kg, vg, posrel, agg, B, M);
  gemm(st, w.lew, agg, w.leb, x, x, B, inC, 64, M, 0);   // idn + conv1d(agg)
}

void sa_knn(hipStream_t st, Arena ar, const SAW& w, const float* xyz,
            const float* pts, int B, int N, int C, int S,
            float* new_xyz, float* out_pts) {
  float* xyzT = ar.f((size_t)B * N * 3);
  L1D(k_perm021, (size_t)B * 3 * N, xyz, xyzT, 3, N);
  int* fidx = ar.i((size_t)B * S);
  k_fps<<<dim3((unsigned)B), 256, 0, st>>>(xyzT, fidx, N, S);
  L1D(k_gather_ch, (size_t)B * 3 * S, xyz, fidx, new_xyz, 3, N, S);
  float* nxT = ar.f((size_t)B * S * 3);
  L1D(k_perm021, (size_t)B * 3 * S, new_xyz, nxT, 3, S);
  int* idx = ar.i((size_t)B * S * 16);
  k_knn<<<dim3((unsigned)cdiv(S, 128), (unsigned)B), 128, 0, st>>>(xyzT, nxT, idx, N, S);
  const int Ci = w.ci;                       // 3 + C
  float* g = ar.f((size_t)B * Ci * S * 16);
  L1D(k_group16, (size_t)B * 3 * S * 16, xyz, idx, new_xyz, g, 3, N, S, Ci, 0);
  L1D(k_group16, (size_t)B * C * S * 16, pts, idx, nullptr, g, C, N, S, Ci, 3);
  float* h1 = ar.f((size_t)B * w.h * S * 16);
  gemm(st, w.w0, g, w.b0, nullptr, h1, B, w.h, Ci, S * 16, 1);
  float* h2 = ar.f((size_t)B * w.o * S * 16);
  gemm(st, w.w1, h1, w.b1, nullptr, h2, B, w.o, w.h, S * 16, 0);
  L1D(k_maxlast, (size_t)B * w.o * S, h2, out_pts, 16);
}

void skip_tr(hipStream_t st, Arena ar, const SkipW& w, const float* pos,
             const float* key_f, const float* query_f, float* H, int B, int n) {
  float* cat = ar.f((size_t)B * 256 * n);
  L1D(k_repeat_ch, (size_t)B * 128 * n, key_f, cat, 128, n, 1, 256, 0);
  L1D(k_repeat_ch, (size_t)B * 128 * n, query_f, cat, 128, n, 1, 256, 128);
  float* val = ar.f((size_t)B * 128 * n);
  mlp_res(st, ar, w.mlp_v, cat, val, B, n);
  float* kf = ar.f((size_t)B * 64 * n);
  float* qf = ar.f((size_t)B * 64 * n);
  float* vf = ar.f((size_t)B * 64 * n);
  gemm(st, w.kw, key_f, w.kb, nullptr, kf, B, 64, 128, n, 0);
  gemm(st, w.qw, query_f, w.qb, nullptr, qf, B, 64, 128, n, 0);
  gemm(st, w.vw, val, w.vb, nullptr, vf, B, 64, 128, n, 0);
  float* posT = ar.f((size_t)B * n * 3);
  L1D(k_perm021, (size_t)B * 3 * n, pos, posT, 3, n);
  int* idx = ar.i((size_t)B * n * 16);
  k_knn<<<dim3((unsigned)cdiv(n, 128), (unsigned)B), 128, 0, st>>>(posT, posT, idx, n, n);
  float* kg = ar.f((size_t)B * 64 * n * 16);
  float* vg = ar.f((size_t)B * 64 * n * 16);
  L1D(k_group16, (size_t)B * 64 * n * 16, kf, idx, nullptr, kg, 64, n, n, 64, 0);
  L1D(k_group16, (size_t)B * 64 * n * 16, vf, idx, nullptr, vg, 64, n, n, 64, 0);
  float* posrel = ar.f((size_t)B * 3 * n * 16);
  L1D(k_group16, (size_t)B * 3 * n * 16, pos, idx, pos, posrel, 3, n, n, 3, 0);
  float* agg = ar.f((size_t)B * 64 * n);
  attn_core(st, ar, w.a, qf, kg, vg, posrel, agg, B, n);
  gemm(st, w.ew, agg, w.eb, val, H, B, 128, 64, n, 0);   // + idn(value)
}

void spd_step(hipStream_t st, Arena ar, const SpdW& w, const float* pcd_prev,
              const float* feat, const float* K_prev, float* pcd_out,
              float* K_out, int B, int n) {
  const int uf = w.uf, nu = n * uf;
  float* f1 = ar.f((size_t)B * 128 * n);
  mlp_conv(st, ar, w.mlp1, pcd_prev, f1, B, n, 0);
  float* gmax = ar.f((size_t)B * 128);
  L1D(k_maxlast, (size_t)B * 128, f1, gmax, n);
  float* f1c = ar.f((size_t)B * 768 * n);
  L1D(k_repeat_ch, (size_t)B * 128 * n, f1, f1c, 128, n, 1, 768, 0);
  L1D(k_repeat_ch, (size_t)B * 128 * n, gmax, f1c, 128, 1, n, 768, 128);
  L1D(k_repeat_ch, (size_t)B * 512 * n, feat, f1c, 512, 1, n, 768, 256);
  float* Q = ar.f((size_t)B * 128 * n);
  mlp_conv(st, ar, w.mlp2, f1c, Q, B, n, 0);
  float* H = ar.f((size_t)B * 128 * n);
  skip_tr(st, ar, w.skip, pcd_prev, K_prev ? K_prev : Q, Q, H, B, n);
  float* fc = ar.f((size_t)B * 32 * n);
  mlp_conv(st, ar, w.mlp_ps, H, fc, B, n, 0);
  float* cat = ar.f((size_t)B * 256 * nu);
  L1D(k_ps_deconv, (size_t)B * 128 * nu, fc, w.psw, cat, n, uf, 256, 0);
  L1D(k_repeat_ch, (size_t)B * 128 * nu, H, cat, 128, n, uf, 256, 128);
  mlp_res(st, ar, w.mdf, cat, K_out, B, nu);
  float* r = ar.f((size_t)B * 128 * nu);
  L1D(k_relu_copy, (size_t)B * 128 * nu, K_out, r);
  float* delta = ar.f((size_t)B * 3 * nu);
  mlp_conv(st, ar, w.md, r, delta, B, nu, 2);            // tanh fused; radius=1
  L1D(k_repeat_add, (size_t)B * 3 * nu, pcd_prev, delta, pcd_out, n, uf);
}

} // namespace

extern "C" void kernel_launch(void* const* d_in, const int* in_sizes, int n_in,
                              void* d_out, int out_size, void* d_ws, size_t ws_size,
                              hipStream_t stream) {
  (void)in_sizes; (void)n_in; (void)out_size;
  hipStream_t st = stream;
  const int B = 4, N = 2048;
  const float* pc = (const float*)d_in[0];       // (B,2048,3)
  // d_in[1], d_in[2] (gt_ones_*) intentionally unused, as in the reference.

  // ---- params: jax pytree flatten order (dicts sorted by key) ----
  int cur = 3;
  auto nxt = [&]() -> const float* { return (const float*)d_in[cur++]; };
  auto rdMRes = [&](int i, int h, int o) {
    MResW m; m.b1 = nxt(); m.b2 = nxt(); m.bs = nxt();
    m.w1 = nxt(); m.w2 = nxt(); m.ws = nxt(); m.ci = i; m.h = h; m.o = o; return m;
  };
  auto rdMConv = [&](int i, int h, int o) {
    MConvW m; m.b0 = nxt(); m.b1 = nxt(); m.w0 = nxt(); m.w1 = nxt();
    m.ci = i; m.h = h; m.o = o; return m;
  };
  // dec.seed: b40,b41,m1,m2,m3,psb,psw,w40,w41
  SeedW seed;
  seed.b40 = nxt(); seed.b41 = nxt();
  seed.m1 = rdMRes(640, 128, 128);
  seed.m2 = rdMRes(128, 64, 128);
  seed.m3 = rdMRes(640, 128, 128);
  seed.psb = nxt(); seed.psw = nxt(); seed.w40 = nxt(); seed.w41 = nxt();
  // dec.spd[i]: md,mdf,mlp1,mlp2,mlp_ps,psw,skip
  auto rdSkip = [&]() {
    SkipW s;
    s.a.ab0 = nxt(); s.a.ab1 = nxt(); s.a.abt0 = nxt(); s.a.ag0 = nxt();
    s.a.aw0 = nxt(); s.a.aw1 = nxt();
    s.eb = nxt(); s.ew = nxt(); s.kb = nxt(); s.kw = nxt();
    s.mlp_v = rdMRes(256, 128, 128);
    s.a.pb0 = nxt(); s.a.pb1 = nxt(); s.a.pbt0 = nxt(); s.a.pg0 = nxt();
    s.a.pw0 = nxt(); s.a.pw1 = nxt();
    s.qb = nxt(); s.qw = nxt(); s.vb = nxt(); s.vw = nxt();
    return s;
  };
  auto rdSpd = [&](int uf) {
    SpdW p;
    p.md     = rdMConv(128, 64, 3);
    p.mdf    = rdMRes(256, 128, 128);
    p.mlp1   = rdMConv(3, 64, 128);
    p.mlp2   = rdMConv(768, 256, 128);
    p.mlp_ps = rdMConv(128, 64, 32);
    p.psw    = nxt();
    p.skip   = rdSkip();
    p.uf     = uf;
    return p;
  };
  SpdW spdw[4] = { rdSpd(1), rdSpd(2), rdSpd(2), rdSpd(2) };
  // enc: sa1,sa2,sa3,t1,t2
  auto rdSA = [&](int ci2, int h, int o) {
    SAW s; s.b0 = nxt(); s.b1 = nxt(); s.w0 = nxt(); s.w1 = nxt();
    s.ci = ci2; s.h = h; s.o = o; return s;
  };
  SAW sa1 = rdSA(6, 64, 128), sa2 = rdSA(131, 128, 256), sa3 = rdSA(259, 512, 512);
  auto rdTrans = [&](int inC) {
    TransW t;
    t.a.ab0 = nxt(); t.a.ab1 = nxt(); t.a.abt0 = nxt(); t.a.ag0 = nxt();
    t.a.aw0 = nxt(); t.a.aw1 = nxt();
    t.kb = nxt(); t.kw = nxt(); t.leb = nxt(); t.lew = nxt();
    t.lsb = nxt(); t.lsw = nxt();
    t.a.pb0 = nxt(); t.a.pb1 = nxt(); t.a.pbt0 = nxt(); t.a.pg0 = nxt();
    t.a.pw0 = nxt(); t.a.pw1 = nxt();
    t.qb = nxt(); t.qw = nxt(); t.vb = nxt(); t.vw = nxt();
    t.inC = inC; return t;
  };
  TransW t1 = rdTrans(128), t2 = rdTrans(256);

  // ---- workspace arena (by-value copies give stack-scoped temporaries) ----
  Arena root{ (char*)d_ws, 0, ws_size };
  float* feat = root.f((size_t)B * 512);
  float* pcdA = root.f((size_t)B * 3 * 4096);
  float* pcdB = root.f((size_t)B * 3 * 4096);
  float* Ka   = root.f((size_t)B * 128 * 4096);
  float* Kb   = root.f((size_t)B * 128 * 4096);

  { // ======================= encoder =======================
    Arena ar = root;
    float* pct = ar.f((size_t)B * 3 * N);
    L1D(k_perm021, (size_t)B * N * 3, pc, pct, N, 3);            // (B,N,3)->(B,3,N)
    float* l1x = ar.f((size_t)B * 3 * 512);
    float* l1p = ar.f((size_t)B * 128 * 512);
    sa_knn(st, ar, sa1, pct, pct, B, N, 3, 512, l1x, l1p);
    transformer(st, ar, t1, l1p, l1x, B, 512);
    float* l2x = ar.f((size_t)B * 3 * 128);
    float* l2p = ar.f((size_t)B * 256 * 128);
    sa_knn(st, ar, sa2, l1x, l1p, B, 512, 128, 128, l2x, l2p);
    transformer(st, ar, t2, l2p, l2x, B, 128);
    // sa_all (group_all)
    float* g = ar.f((size_t)B * 259 * 128);
    L1D(k_repeat_ch, (size_t)B * 3 * 128, l2x, g, 3, 128, 1, 259, 0);
    L1D(k_repeat_ch, (size_t)B * 256 * 128, l2p, g, 256, 128, 1, 259, 3);
    float* h1 = ar.f((size_t)B * 512 * 128);
    gemm(st, sa3.w0, g, sa3.b0, nullptr, h1, B, 512, 259, 128, 1);
    float* h2 = ar.f((size_t)B * 512 * 128);
    gemm(st, sa3.w1, h1, sa3.b1, nullptr, h2, B, 512, 512, 128, 0);
    L1D(k_maxlast, (size_t)B * 512, h2, feat, 128);
  }

  { // ======================= decoder =======================
    Arena ar = root;
    // seed generator: ConvTranspose1d(512,128,256) == feat(B,512) @ psw(512, 128*256)
    float* x1 = ar.f((size_t)B * 128 * 256);
    gemm(st, feat, seed.psw, nullptr, nullptr, x1, 1, B, 512, 128 * 256, 0);
    L1D(k_bias_ch, (size_t)B * 128 * 256, x1, seed.psb, 128, 256);
    float* cat = ar.f((size_t)B * 640 * 256);
    L1D(k_repeat_ch, (size_t)B * 128 * 256, x1, cat, 128, 256, 1, 640, 0);
    L1D(k_repeat_ch, (size_t)B * 512 * 256, feat, cat, 512, 1, 256, 640, 128);
    float* xa = ar.f((size_t)B * 128 * 256);
    mlp_res(st, ar, seed.m1, cat, xa, B, 256);
    float* xb = ar.f((size_t)B * 128 * 256);
    mlp_res(st, ar, seed.m2, xa, xb, B, 256);
    L1D(k_repeat_ch, (size_t)B * 128 * 256, xb, cat, 128, 256, 1, 640, 0);
    mlp_res(st, ar, seed.m3, cat, xa, B, 256);
    float* t64 = ar.f((size_t)B * 64 * 256);
    gemm(st, seed.w40, xa, seed.b40, nullptr, t64, B, 64, 128, 256, 1);
    float* sc = ar.f((size_t)B * 3 * 256);
    gemm(st, seed.w41, t64, seed.b41, nullptr, sc, B, 3, 64, 256, 0);
    // merge coarse(256) + partial(2048) rows, fps_subsample to 512
    float* coarse = ar.f((size_t)B * 256 * 3);
    L1D(k_perm021, (size_t)B * 3 * 256, sc, coarse, 3, 256);
    float* merged = ar.f((size_t)B * 2304 * 3);
    L1D(k_copy_rows3, (size_t)B * 256 * 3, coarse, merged, 256, 2304, 0);
    L1D(k_copy_rows3, (size_t)B * 2048 * 3, pc, merged, 2048, 2304, 256);
    int* fi = ar.i((size_t)B * 512);
    k_fps<<<dim3((unsigned)B), 256, 0, st>>>(merged, fi, 2304, 512);
    float* p0 = ar.f((size_t)B * 512 * 3);
    L1D(k_gather_rows3, (size_t)B * 512 * 3, merged, fi, p0, 2304, 512);
    L1D(k_perm021, (size_t)B * 512 * 3, p0, pcdA, 512, 3);       // (B,3,512)
    // SPD chain: up_factors [1,2,2,2] -> 512 .. 4096 points
    int n = 512;
    const float* Kprev = nullptr;
    float* pin = pcdA;
    for (int i = 0; i < 4; ++i) {
      float* po = (i & 1) ? pcdA : pcdB;
      float* ko = (i & 1) ? Kb : Ka;
      spd_step(st, ar, spdw[i], pin, feat, Kprev, po, ko, B, n);
      n *= spdw[i].uf;
      pin = po; Kprev = ko;
    }
    L1D(k_perm021, (size_t)B * 3 * 4096, pin, (float*)d_out, 3, 4096);  // (B,4096,3)
  }
}
